// LowDimQKMultiHeadAttention_9337258901813
// MI455X (gfx1250) — matrix operand order repaired and checked by reading, not compiled, not run on hardware
//
#include <hip/hip_runtime.h>
#include <hip/hip_bf16.h>

// ---------------------------------------------------------------------------
// LowDimQKMultiHeadAttention for MI455X (gfx1250, wave32, WMMA bf16 pipeline)
//
//  B=4, TQ=TK=2048, DQ=DKV=1024, DV=256, QK_DIM=128, H=8, HD=16, OUT=1024
//
//  k0: V^T -> bf16 ws  (4, 256, 2048)                (one-time transpose)
//  k1: Q_low = 0.25*(Q @ Wq^T + bq)  -> bf16 ws      (scale folded in)
//  k1: K_low =       (K @ Wk^T + bk) -> bf16 ws
//  k2: flash attention per (b,h,qtile): S^T via WMMA, online softmax,
//      P (registers) @ V via WMMA; V chunks double-buffered into LDS with
//      global_load_async_to_lds_b128 (ASYNCcnt) -> bf16 ws (B*TQ, 2048)
//  k3: out = AO @ Wo^T + bo -> fp32 d_out
// ---------------------------------------------------------------------------

typedef __bf16 bf16;
typedef bf16  bf16x8 __attribute__((ext_vector_type(8)));
typedef bf16  v16bf  __attribute__((ext_vector_type(16)));
typedef float v8f    __attribute__((ext_vector_type(8)));

#define B_    4
#define TQ_   2048
#define TK_   2048
#define DQ_   1024
#define DKV_  1024
#define DV_   256
#define QKD_  128
#define H_    8
#define HD_   16
#define ODIM_ 1024

static __device__ __forceinline__ v8f wmma_bf(v16bf a, v16bf b, v8f c) {
  // D = A(16x32 bf16) x B(32x16 bf16) + C(16x16 f32)
  return __builtin_amdgcn_wmma_f32_16x16x32_bf16(false, a, false, b, (short)0, c,
                                                 false, false);
}
static __device__ __forceinline__ v16bf cat8(bf16x8 a, bf16x8 b) {
  return __builtin_shufflevector(a, b, 0, 1, 2, 3, 4, 5, 6, 7, 8, 9, 10, 11, 12,
                                 13, 14, 15);
}

// ---------------------------------------------------------------------------
// k0: Vt[b][c][k] = (bf16)V[b][k][c]; grid (TK/64, DV/64, B), block 256
// ---------------------------------------------------------------------------
__global__ __launch_bounds__(256) void transpose_v(const float* __restrict__ V,
                                                   bf16* __restrict__ Vt) {
  __shared__ __align__(16) bf16 tile[64][72];  // [col][key], 144B row stride
  const int tid = threadIdx.x;
  const int k0 = blockIdx.x * 64, c0 = blockIdx.y * 64, b = blockIdx.z;
  float4 f[4];
#pragma unroll
  for (int i = 0; i < 4; ++i) {  // 64 keys x 16 float4-groups = 1024 slots
    int idx = tid + i * 256;
    int key = idx >> 4, grp = idx & 15;
    f[i] = *(const float4*)(V + ((size_t)b * TK_ + k0 + key) * DV_ + c0 + grp * 4);
  }
#pragma unroll
  for (int i = 0; i < 4; ++i) {
    int idx = tid + i * 256;
    int key = idx >> 4, grp = idx & 15;
    tile[grp * 4 + 0][key] = (bf16)f[i].x;
    tile[grp * 4 + 1][key] = (bf16)f[i].y;
    tile[grp * 4 + 2][key] = (bf16)f[i].z;
    tile[grp * 4 + 3][key] = (bf16)f[i].w;
  }
  __syncthreads();
#pragma unroll
  for (int i = 0; i < 2; ++i) {  // 64 cols x 8 b128-segments = 512 slots
    int idx = tid + i * 256;
    int col = idx >> 3, seg = idx & 7;
    *(bf16x8*)(Vt + ((size_t)b * DV_ + c0 + col) * TK_ + k0 + seg * 8) =
        *(const bf16x8*)&tile[col][seg * 8];
  }
}

// ---------------------------------------------------------------------------
// k1: C[M,N](bf16) = scale * (A[M,Kd](f32) @ W[N,Kd](f32)^T + bias[N]); N==128
// block = 256 threads (8 waves); block tile = 128(M) x 128(N); k-step 32
// ---------------------------------------------------------------------------
__global__ __launch_bounds__(256) void proj_gemm(
    const float* __restrict__ A, const float* __restrict__ W,
    const float* __restrict__ bias, bf16* __restrict__ C, int M, int Kd, int N,
    float scale) {
  __shared__ __align__(16) bf16 As[128][40];
  __shared__ __align__(16) bf16 Ws[128][40];
  const int tid = threadIdx.x;
  const int lane = tid & 31, w = tid >> 5;
  const int r = lane & 15, half = lane >> 4;
  const int base0 = half * 8;    // A-fragment K interleave
  const int koff = half * 16;    // unused; B fragment also uses base0
  const int mbase = blockIdx.x * 128;

  v8f acc[8];
#pragma unroll
  for (int t = 0; t < 8; ++t)
#pragma unroll
    for (int v = 0; v < 8; ++v) acc[t][v] = 0.f;

  for (int kc = 0; kc < Kd; kc += 32) {
    __syncthreads();
    // ---- load phase: 8 b128 loads in flight
    float4 fa[4], fg[4];
#pragma unroll
    for (int i = 0; i < 4; ++i) {  // 128 rows x 8 float4-groups = 1024 slots
      int idx = tid + i * 256;
      int row = idx >> 3, grp = idx & 7;
      fa[i] = *(const float4*)(A + (size_t)(mbase + row) * Kd + kc + grp * 4);
      fg[i] = *(const float4*)(W + (size_t)row * Kd + kc + grp * 4);
    }
    // ---- convert + LDS store phase
#pragma unroll
    for (int i = 0; i < 4; ++i) {
      int idx = tid + i * 256;
      int row = idx >> 3, grp = idx & 7;
      bf16* p = &As[row][grp * 4];
      p[0] = (bf16)fa[i].x; p[1] = (bf16)fa[i].y;
      p[2] = (bf16)fa[i].z; p[3] = (bf16)fa[i].w;
      bf16* q = &Ws[row][grp * 4];
      q[0] = (bf16)fg[i].x; q[1] = (bf16)fg[i].y;
      q[2] = (bf16)fg[i].z; q[3] = (bf16)fg[i].w;
    }
    __syncthreads();
    // ---- fragments + WMMA chain
    bf16x8 alo = *(const bf16x8*)&As[w * 16 + r][base0];
    bf16x8 ahi = *(const bf16x8*)&As[w * 16 + r][16 + base0];
    v16bf af = cat8(alo, ahi);
    v16bf bfr[8];
#pragma unroll
    for (int nt = 0; nt < 8; ++nt) {
      bf16x8 blo = *(const bf16x8*)&Ws[nt * 16 + r][base0];
      bf16x8 bhi = *(const bf16x8*)&Ws[nt * 16 + r][16 + base0];
      bfr[nt] = cat8(blo, bhi);
    }
#pragma unroll
    for (int nt = 0; nt < 8; ++nt) acc[nt] = wmma_bf(af, bfr[nt], acc[nt]);
  }
#pragma unroll
  for (int nt = 0; nt < 8; ++nt) {
    int col = nt * 16 + r;
    float bb = bias[col];
    size_t rowb = (size_t)mbase + w * 16 + half * 8;
#pragma unroll
    for (int v = 0; v < 8; ++v)
      C[(rowb + v) * N + col] = (bf16)(scale * (acc[nt][v] + bb));
  }
}

// ---------------------------------------------------------------------------
// k2: flash attention. grid = (TQ/128, H, B), block = 256 (8 waves).
// Each wave owns 16 query rows; O accumulator = 16 x 256 f32 (16 WMMA C tiles).
// V^T chunks stream into a double-buffered LDS tile via async copy.
// ---------------------------------------------------------------------------
__global__ __launch_bounds__(256) void flash_attn(
    const bf16* __restrict__ Qlow, const bf16* __restrict__ Klow,
    const bf16* __restrict__ Vt, const unsigned char* __restrict__ mask,
    bf16* __restrict__ AO) {
  __shared__ __align__(16) bf16 Vl[2][256][40];  // [buf][vcol][key], 80B rows
  __shared__ float fbuf[8][16];                  // per-wave broadcast bounce
  const int tid = threadIdx.x, lane = tid & 31, w = tid >> 5;
  const int r = lane & 15, half = lane >> 4;
  const int base0 = half * 8, koff = half * 16;
  const int h = blockIdx.y, b = blockIdx.z;
  const size_t rowsB = (size_t)b * TK_;
  const int q0 = blockIdx.x * 128 + w * 16;
  const bf16* Vtg = Vt + (size_t)b * DV_ * TK_;  // [vcol][key]

  bf16x8 z8;
#pragma unroll
  for (int i = 0; i < 8; ++i) z8[i] = (bf16)0.f;

  // Q as the B operand of S^T = K_chunk @ Q^T. WMMA-K = head dim (16, padded).
  v16bf qb;
#pragma unroll
  for (int i = 0; i < 16; ++i) qb[i] = (bf16)0.f;
  {
    const bf16* qp = Qlow + ((size_t)b * TQ_ + q0 + r) * QKD_ + h * HD_ + base0;
    qb = cat8(*(const bf16x8*)qp, z8);
  }

  v8f o[16];
#pragma unroll
  for (int t = 0; t < 16; ++t)
#pragma unroll
    for (int v = 0; v < 8; ++v) o[t][v] = 0.f;
  float mrun = -1e30f, lrun = 0.f;

  // prefetch: 4 x b128 async copies per thread = 16KB (32 keys x 256 cols bf16)
  // slot = vcol*4 + seg;  LDS row stride 80B, segs at 0/16/32/48B
#define PREFETCH_V(BUF, KK)                                                     \
  {                                                                             \
    _Pragma("unroll") for (int i = 0; i < 4; ++i) {                             \
      int idx = tid + i * 256;                                                  \
      int row = idx >> 2, seg = idx & 3;                                        \
      unsigned long long ga = (unsigned long long)(size_t)(                     \
          Vtg + (size_t)row * TK_ + (KK) + seg * 8);                            \
      unsigned la = (unsigned)(size_t)&Vl[BUF][row][seg * 8];                   \
      asm volatile("global_load_async_to_lds_b128 %0, %1, off" ::"v"(la),       \
                   "v"(ga)                                                      \
                   : "memory");                                                 \
    }                                                                           \
  }

  PREFETCH_V(0, 0)

  for (int kk = 0; kk < TK_; kk += 32) {
    const int cur = (kk >> 5) & 1;
    __syncthreads();  // all waves done reading Vl[cur^1] (prev compute)
    if (kk + 32 < TK_) {
      PREFETCH_V(cur ^ 1, kk + 32)
      asm volatile("s_wait_asynccnt 0x4" ::: "memory");  // older 4 (cur) done
    } else {
      asm volatile("s_wait_asynccnt 0x0" ::: "memory");
    }
    __syncthreads();  // Vl[cur] complete for every wave

    // ---- S^T (keys x queries): two 16x16 tiles, A = K_low rows (padded K)
    const bf16* kp = Klow + (rowsB + kk + r) * QKD_ + h * HD_ + base0;
    bf16x8 k0 = *(const bf16x8*)kp;
    bf16x8 k1 = *(const bf16x8*)(kp + 16 * QKD_);
    const unsigned char* mp = mask + (size_t)b * TK_ + kk + half * 8;
    unsigned long long mm0 = *(const unsigned long long*)(mp);
    unsigned long long mm1 = *(const unsigned long long*)(mp + 16);
    v16bf ka0 = cat8(k0, z8);
    v16bf ka1 = cat8(k1, z8);
    v8f zc;
#pragma unroll
    for (int v = 0; v < 8; ++v) zc[v] = 0.f;
    v8f s0 = wmma_bf(ka0, qb, zc);  // keys kk+half*8+v
    v8f s1 = wmma_bf(ka1, qb, zc);  // keys kk+16+half*8+v

    // ---- key padding mask (scale 1/sqrt(HD) already folded into Qlow)
#pragma unroll
    for (int v = 0; v < 8; ++v) {
      if ((mm0 >> (8 * v)) & 0xffull) s0[v] = -1e30f;
      if ((mm1 >> (8 * v)) & 0xffull) s1[v] = -1e30f;
    }

    // ---- online softmax; per-query stats live on lanes q and q+16 (synced)
    float cm = -1e30f;
#pragma unroll
    for (int v = 0; v < 8; ++v) cm = fmaxf(cm, fmaxf(s0[v], s1[v]));
    cm = fmaxf(cm, __shfl_xor(cm, 16, 32));
    float mn = fmaxf(mrun, cm);
    float fac = __expf(mrun - mn);
    float ps = 0.f;
    v8f p0, p1;
#pragma unroll
    for (int v = 0; v < 8; ++v) {
      p0[v] = __expf(s0[v] - mn);
      p1[v] = __expf(s1[v] - mn);
      ps += p0[v] + p1[v];
    }
    ps += __shfl_xor(ps, 16, 32);
    lrun = lrun * fac + ps;
    mrun = mn;

    // ---- P C-layout == A-fragment layout for 16x16x32 WMMA: just pack bf16
    v16bf pa;
#pragma unroll
    for (int v = 0; v < 8; ++v) {
      pa[v] = (bf16)p0[v];
      pa[8 + v] = (bf16)p1[v];
    }

    // ---- rescale O: broadcast per-query factor via per-wave LDS bounce
    if (half == 0) fbuf[w][r] = fac;
    asm volatile("" ::: "memory");  // keep store before loads; LDS is in-order
    float fq[8];
#pragma unroll
    for (int v = 0; v < 8; ++v) fq[v] = fbuf[w][v + half * 8];
#pragma unroll
    for (int t = 0; t < 16; ++t)
#pragma unroll
      for (int v = 0; v < 8; ++v) o[t][v] *= fq[v];

    // ---- O += P @ V : 16 tiles across DV=256 (fragments grouped by 4)
#pragma unroll
    for (int g = 0; g < 4; ++g) {
      v16bf vb[4];
#pragma unroll
      for (int t2 = 0; t2 < 4; ++t2) {
        int t = g * 4 + t2;
        bf16x8 v0 = *(const bf16x8*)&Vl[cur][t * 16 + r][base0];
        bf16x8 v1 = *(const bf16x8*)&Vl[cur][t * 16 + r][16 + base0];
        vb[t2] = cat8(v0, v1);
      }
#pragma unroll
      for (int t2 = 0; t2 < 4; ++t2)
        o[g * 4 + t2] = wmma_bf(pa, vb[t2], o[g * 4 + t2]);
    }
  }
#undef PREFETCH_V

  // ---- finalize: divide by l (broadcast 1/l like the rescale factor)
  if (half == 0) fbuf[w][r] = 1.f / lrun;
  asm volatile("" ::: "memory");
  float iq[8];
#pragma unroll
  for (int v = 0; v < 8; ++v) iq[v] = fbuf[w][v + half * 8];
  bf16* op = AO + ((size_t)b * TQ_ + q0) * (H_ * DV_) + h * DV_;
#pragma unroll
  for (int t = 0; t < 16; ++t) {
    int col = t * 16 + r;
#pragma unroll
    for (int v = 0; v < 8; ++v)
      op[(size_t)(v + half * 8) * (H_ * DV_) + col] = (bf16)(o[t][v] * iq[v]);
  }
}

// ---------------------------------------------------------------------------
// k3: Out[M,1024](f32) = A[M,2048](bf16) @ Wo[1024,2048](f32)^T + bo
// grid = (M/128, N/128)
// ---------------------------------------------------------------------------
__global__ __launch_bounds__(256) void out_gemm(const bf16* __restrict__ A,
                                                const float* __restrict__ W,
                                                const float* __restrict__ bias,
                                                float* __restrict__ C) {
  __shared__ __align__(16) bf16 As[128][40];
  __shared__ __align__(16) bf16 Ws[128][40];
  const int tid = threadIdx.x;
  const int lane = tid & 31, w = tid >> 5;
  const int r = lane & 15, half = lane >> 4;
  const int base0 = half * 8, koff = half * 16;
  const int mbase = blockIdx.x * 128;
  const int nbase = blockIdx.y * 128;

  v8f acc[8];
#pragma unroll
  for (int t = 0; t < 8; ++t)
#pragma unroll
    for (int v = 0; v < 8; ++v) acc[t][v] = 0.f;

  for (int kc = 0; kc < H_ * DV_; kc += 32) {
    __syncthreads();
    // ---- load phase
    bf16x8 ba[2];
    float4 fg[4];
#pragma unroll
    for (int i = 0; i < 2; ++i) {  // A tile (bf16): 128 rows x 4 groups of 8
      int idx = tid + i * 256;
      int row = idx >> 2, grp = idx & 3;
      ba[i] =
          *(const bf16x8*)(A + (size_t)(mbase + row) * (H_ * DV_) + kc + grp * 8);
    }
#pragma unroll
    for (int i = 0; i < 4; ++i) {  // W tile (f32)
      int idx = tid + i * 256;
      int row = idx >> 3, grp = idx & 7;
      fg[i] =
          *(const float4*)(W + (size_t)(nbase + row) * (H_ * DV_) + kc + grp * 4);
    }
    // ---- store/convert phase
#pragma unroll
    for (int i = 0; i < 2; ++i) {
      int idx = tid + i * 256;
      int row = idx >> 2, grp = idx & 3;
      *(bf16x8*)&As[row][grp * 8] = ba[i];
    }
#pragma unroll
    for (int i = 0; i < 4; ++i) {
      int idx = tid + i * 256;
      int row = idx >> 3, grp = idx & 7;
      bf16* q = &Ws[row][grp * 4];
      q[0] = (bf16)fg[i].x; q[1] = (bf16)fg[i].y;
      q[2] = (bf16)fg[i].z; q[3] = (bf16)fg[i].w;
    }
    __syncthreads();
    // ---- fragments + WMMA chain
    bf16x8 alo = *(const bf16x8*)&As[w * 16 + r][base0];
    bf16x8 ahi = *(const bf16x8*)&As[w * 16 + r][16 + base0];
    v16bf af = cat8(alo, ahi);
    v16bf bfr[8];
#pragma unroll
    for (int nt = 0; nt < 8; ++nt) {
      bf16x8 blo = *(const bf16x8*)&Ws[nt * 16 + r][base0];
      bf16x8 bhi = *(const bf16x8*)&Ws[nt * 16 + r][16 + base0];
      bfr[nt] = cat8(blo, bhi);
    }
#pragma unroll
    for (int nt = 0; nt < 8; ++nt) acc[nt] = wmma_bf(af, bfr[nt], acc[nt]);
  }
#pragma unroll
  for (int nt = 0; nt < 8; ++nt) {
    int col = nbase + nt * 16 + r;
    float bb = bias[col];
    size_t rowb = (size_t)mbase + w * 16 + half * 8;
#pragma unroll
    for (int v = 0; v < 8; ++v) C[(rowb + v) * ODIM_ + col] = acc[nt][v] + bb;
  }
}

// ---------------------------------------------------------------------------
extern "C" void kernel_launch(void* const* d_in, const int* in_sizes, int n_in,
                              void* d_out, int out_size, void* d_ws,
                              size_t ws_size, hipStream_t stream) {
  (void)in_sizes; (void)n_in; (void)out_size; (void)ws_size;
  const float* Q = (const float*)d_in[0];
  const float* K = (const float*)d_in[1];
  const float* V = (const float*)d_in[2];
  const unsigned char* mask = (const unsigned char*)d_in[3];  // jnp.bool_ bytes
  const float* Wq = (const float*)d_in[4];
  const float* bq = (const float*)d_in[5];
  const float* Wk = (const float*)d_in[6];
  const float* bk = (const float*)d_in[7];
  const float* Wo = (const float*)d_in[8];
  const float* bo = (const float*)d_in[9];

  // workspace: Qlow 2MB | Klow 2MB | AO 32MB | Vt 4MB (all bf16)
  bf16* Qlow = (bf16*)d_ws;
  bf16* Klow = Qlow + (size_t)B_ * TQ_ * QKD_;
  bf16* AO = Klow + (size_t)B_ * TK_ * QKD_;
  bf16* Vtw = AO + (size_t)B_ * TQ_ * (H_ * DV_);

  transpose_v<<<dim3(TK_ / 64, DV_ / 64, B_), 256, 0, stream>>>(V, Vtw);
  // fold softmax scale 1/sqrt(HD)/TEMPERATURE = 0.25 into Q_low
  proj_gemm<<<dim3((B_ * TQ_) / 128), 256, 0, stream>>>(Q, Wq, bq, Qlow,
                                                        B_ * TQ_, DQ_, QKD_,
                                                        0.25f);
  proj_gemm<<<dim3((B_ * TK_) / 128), 256, 0, stream>>>(K, Wk, bk, Klow,
                                                        B_ * TK_, DKV_, QKD_,
                                                        1.0f);
  flash_attn<<<dim3(TQ_ / 128, H_, B_), 256, 0, stream>>>(Qlow, Klow, Vtw, mask,
                                                          AO);
  out_gemm<<<dim3((B_ * TQ_) / 128, ODIM_ / 128), 256, 0, stream>>>(
      AO, Wo, bo, (float*)d_out);
}
